// DDI_decagon_84482006713338
// MI455X (gfx1250) — matrix operand order repaired and checked by reading, not compiled, not run on hardware
//
#include <hip/hip_runtime.h>
#include <hip/hip_bf16.h>

// ---------------------------------------------------------------------------
// MI455X (gfx1250) implementation of 2-layer RGCN + DEDICOM decoder.
// wave32; WMMA f32_16x16x32_f16 for all GEMM work; Tensor Data Mover stages
// the shared 32KB f16 B-tiles into LDS once per workgroup; f32 global atomics
// for the edge scatter; LDS ds_add_f32 for decoder reductions.
// ---------------------------------------------------------------------------

typedef __attribute__((ext_vector_type(16))) _Float16     v16h;
typedef __attribute__((ext_vector_type(8)))  float        v8f;
typedef __attribute__((ext_vector_type(4)))  unsigned int u32x4;
typedef __attribute__((ext_vector_type(8)))  int          i32x8;
typedef __attribute__((ext_vector_type(4)))  int          i32x4;

constexpr int NN = 20000;    // nodes
constexpr int NE = 1000000;  // edges
constexpr int RR = 32;       // relations (== out dim of decoder)
constexpr int HH = 128;      // hidden = in dim
constexpr int NB = 8;        // bases
constexpr int NT = 20000;    // target edges

#if defined(__HIP_DEVICE_COMPILE__) && __has_builtin(__builtin_amdgcn_tensor_load_to_lds) && __has_builtin(__builtin_amdgcn_s_wait_tensorcnt)
#define USE_TDM 1
#else
#define USE_TDM 0
#endif

// ---------------------------------------------------------------------------
// WMMA helpers (layouts per CDNA5 ISA 7.12.2)
// ---------------------------------------------------------------------------
__device__ __forceinline__ v8f wmma_f16(v16h a, v16h b, v8f c) {
  return __builtin_amdgcn_wmma_f32_16x16x32_f16(
      /*neg_a=*/false, a, /*neg_b=*/false, b,
      /*c_mod=*/(short)0, c, /*reuse_a=*/false, /*reuse_b=*/false);
}

// B fragment: Bt stored N-major ([n][k], k contiguous). Lane (half,nn) needs
// K = (t>>3)*16 + half*8 + (t&7) for column n -> two contiguous 8-half runs.
__device__ __forceinline__ v16h load_b_frag(const _Float16* p) {
  v16h b;
#pragma unroll
  for (int t = 0; t < 16; ++t) b[t] = p[(t >> 3) * 16 + (t & 7)];
  return b;
}

// A fragment from an f32 row (convert to f16).
// element t: K = (t>>3)*16 + half*8 + (t&7) -> two contiguous 8-float runs.
__device__ __forceinline__ v16h load_a_frag_f32(const float* row, int k0, int half) {
  v16h a;
#pragma unroll
  for (int part = 0; part < 2; ++part) {
    const float4* p = (const float4*)(row + k0 + part * 16 + half * 8);
    float4 x0 = p[0], x1 = p[1];
    int t = part * 8;
    a[t+0]=(_Float16)x0.x; a[t+1]=(_Float16)x0.y; a[t+2]=(_Float16)x0.z; a[t+3]=(_Float16)x0.w;
    a[t+4]=(_Float16)x1.x; a[t+5]=(_Float16)x1.y; a[t+6]=(_Float16)x1.z; a[t+7]=(_Float16)x1.w;
  }
  return a;
}

// ---------------------------------------------------------------------------
// Issue one Tensor Data Mover load of a 128x128 f16 tile (32KB) into LDS.
// D# per CDNA5 ISA ch.8 (2D tensor, data_size=2B, tile 128x128, stride 128).
// Caller: one wave issues this, then s_wait_tensorcnt + block barrier.
// ---------------------------------------------------------------------------
__device__ __forceinline__ void tdm_issue_128x128_f16(const _Float16* __restrict__ gsrc,
                                                      _Float16* __restrict__ lds_dst) {
#if USE_TDM
  unsigned long long ga = (unsigned long long)(size_t)gsrc;
  unsigned int lds_addr = (unsigned int)(size_t)(void*)lds_dst;  // LDS byte offset
  u32x4 g0;
  g0[0] = 1u;                                   // count=1, user descriptor
  g0[1] = lds_addr;                             // lds_addr[31:0]
  g0[2] = (unsigned int)ga;                     // global_addr[31:0]
  g0[3] = (unsigned int)((ga >> 32) & 0x1FFFFFFu) | (2u << 30);  // addr[56:32], type=2
  i32x8 g1;
  g1[0] = (1 << 16);          // workgroup_mask=0, data_size=1 (2 bytes)
  g1[1] = (HH << 16);         // tensor_dim0[15:0]=128  (w1[31:16])
  g1[2] = (HH << 16);         // tensor_dim0[31:16]=0, tensor_dim1[15:0]=128
  g1[3] = (HH << 16);         // tensor_dim1[31:16]=0, tile_dim0=128
  g1[4] = HH;                 // tile_dim1=128, tile_dim2=0
  g1[5] = HH;                 // tensor_dim0_stride[31:0]=128
  g1[6] = 0;                  // stride0[47:32]=0, tensor_dim1_stride lo=0
  g1[7] = 0;
  i32x4 z4; z4[0] = 0; z4[1] = 0; z4[2] = 0; z4[3] = 0;
  i32x8 z8;
#pragma unroll
  for (int i = 0; i < 8; ++i) z8[i] = 0;
  // clang-23 / therock-10 form: (g0, g1, g2, g3, extra, cpol)
  __builtin_amdgcn_tensor_load_to_lds(g0, g1, z4, z4, z8, 0);
  __builtin_amdgcn_s_wait_tensorcnt(0);
#else
  (void)gsrc; (void)lds_dst;
#endif
}

// Stage a 128x128 f16 tile into LDS, shared by the whole block.
// Must be called by ALL threads of the block (contains a barrier).
__device__ __forceinline__ void stage_tile_128x128_f16(const _Float16* __restrict__ gsrc,
                                                       _Float16* __restrict__ lds_dst) {
#if USE_TDM
  if (threadIdx.x < 32) {  // wave 0 issues a single TDM descriptor
    tdm_issue_128x128_f16(gsrc, lds_dst);
  }
  __syncthreads();
#else
  // fallback: cooperative copy
  const float4* src = (const float4*)gsrc;   // 8 halves per float4
  float4* dst = (float4*)lds_dst;
  for (int i = threadIdx.x; i < 2048; i += blockDim.x) dst[i] = src[i];
  __syncthreads();
#endif
}

// ---------------------------------------------------------------------------
// 1) etype = argmax(edge_attr); relation histogram; per-(dst,rel) edge counts
// ---------------------------------------------------------------------------
__global__ void k_etype(const float* __restrict__ attr,
                        const long long* __restrict__ eidx,
                        int* __restrict__ etype, int* __restrict__ bcnt,
                        int* __restrict__ cnt) {
  int e = blockIdx.x * blockDim.x + threadIdx.x;
  if (e >= NE) return;
  const float* a = attr + (size_t)e * RR;
  int best = 0; float bv = a[0];
#pragma unroll
  for (int r = 1; r < RR; ++r) { float v = a[r]; if (v > bv) { bv = v; best = r; } }
  etype[e] = best;
  atomicAdd(&bcnt[best], 1);
  int dst = (int)eidx[NE + e];
  atomicAdd(&cnt[dst * RR + best], 1);
}

// 2) tiny serial exclusive scan over 32 relation counts
__global__ void k_scan(const int* __restrict__ bcnt, int* __restrict__ boff) {
  if (blockIdx.x == 0 && threadIdx.x == 0) {
    int s = 0; boff[0] = 0;
    for (int i = 0; i < RR; ++i) { s += bcnt[i]; boff[i + 1] = s; }
  }
}

// 3) scatter edge ids into relation buckets (counting sort)
__global__ void k_scatter(const int* __restrict__ etype, const int* __restrict__ boff,
                          int* __restrict__ bfill, int* __restrict__ bucket) {
  int e = blockIdx.x * blockDim.x + threadIdx.x;
  if (e >= NE) return;
  int r = etype[e];
  int p = atomicAdd(&bfill[r], 1);
  bucket[boff[r] + p] = e;
}

// 4) W[r] = sum_b comp[r,b]*bases[b]; stored transposed+f16: Wt[r][o][i]
__global__ void k_expand_w(const float* __restrict__ comp, const float* __restrict__ bases,
                           _Float16* __restrict__ Wt) {
  int idx = blockIdx.x * blockDim.x + threadIdx.x;
  if (idx >= RR * HH * HH) return;
  int r = idx >> 14, rem = idx & 16383;
  int o = rem >> 7, i = rem & 127;
  float acc = 0.f;
#pragma unroll
  for (int b = 0; b < NB; ++b)
    acc += comp[r * NB + b] * bases[((size_t)b * HH + i) * HH + o];
  Wt[((size_t)r * HH + o) * HH + i] = (_Float16)acc;
}

// 5) 128x128 f32 -> f16 (optionally transposed to N-major)
__global__ void k_cvt(const float* __restrict__ A, _Float16* __restrict__ Bt, int transpose) {
  int idx = blockIdx.x * blockDim.x + threadIdx.x;
  if (idx >= HH * HH) return;
  int rrow = idx >> 7, ccol = idx & 127;
  float v = transpose ? A[ccol * HH + rrow] : A[idx];
  Bt[idx] = (_Float16)v;
}

// ---------------------------------------------------------------------------
// 6) Dense root GEMM: Y[M,128] = X[M,128] @ rootT + bias   (init accumulator)
//    8 waves / block; B tile staged in LDS by TDM; 32 WMMAs per wave.
// ---------------------------------------------------------------------------
__global__ __launch_bounds__(256) void k_dense_gemm(
    const float* __restrict__ X, const _Float16* __restrict__ Bt,
    const float* __restrict__ bias, float* __restrict__ Y, int M) {
  __shared__ __align__(16) _Float16 s_w[HH * HH];

  stage_tile_128x128_f16(Bt, s_w);

  int wave = threadIdx.x >> 5, lane = threadIdx.x & 31;
  int half = lane >> 4, nn = lane & 15;
  int mbase = blockIdx.x * 128 + wave * 16;
  if (mbase >= M) return;  // after barrier: safe

  v8f acc[8];
#pragma unroll
  for (int nt = 0; nt < 8; ++nt) {
    float bv = bias[nt * 16 + nn];
#pragma unroll
    for (int j = 0; j < 8; ++j) acc[nt][j] = bv;
  }
  int mrow = mbase + nn;
  if (mrow >= M) mrow = M - 1;
  const float* row = X + (size_t)mrow * HH;
#pragma unroll
  for (int ks = 0; ks < 4; ++ks) {
    int k0 = ks * 32;
    v16h a = load_a_frag_f32(row, k0, half);
#pragma unroll
    for (int nt = 0; nt < 8; ++nt) {
      v16h b = load_b_frag(s_w + (size_t)(nt * 16 + nn) * HH + k0 + half * 8);
      acc[nt] = wmma_f16(a, b, acc[nt]);
    }
  }
#pragma unroll
  for (int nt = 0; nt < 8; ++nt) {
    int ng = nt * 16 + nn;
#pragma unroll
    for (int j = 0; j < 8; ++j) {
      int m = mbase + half * 8 + j;
      if (m < M) Y[(size_t)m * HH + ng] = acc[nt][j];
    }
  }
}

// ---------------------------------------------------------------------------
// 7) Relation message GEMM: for each 16-edge tile of relation r:
//    Y[dst_e] += (1/cnt[dst_e,r]) * X[src_e] @ W_r      (f32 global atomics)
// ---------------------------------------------------------------------------
__global__ __launch_bounds__(256) void k_rel_gemm(
    const float* __restrict__ X, const _Float16* __restrict__ Wt,
    const long long* __restrict__ eidx, const int* __restrict__ bucket,
    const int* __restrict__ boff, const int* __restrict__ cnt,
    float* __restrict__ Y) {
  __shared__ __align__(16) _Float16 s_w[HH * HH];
  __shared__ __align__(16) int   s_src[8][16];
  __shared__ __align__(16) int   s_dst[8][16];
  __shared__ __align__(16) float s_scl[8][16];

  int wave = threadIdx.x >> 5, lane = threadIdx.x & 31;
  int r = blockIdx.y;
  int off = boff[r], nEr = boff[r + 1] - off;
  if (blockIdx.x * 128 >= nEr) return;  // block-uniform early exit (no barrier yet)

  int base = (blockIdx.x * 8 + wave) * 16;
  bool active = (base < nEr);
  int half = lane >> 4, nn = lane & 15;

  if (active && lane < 16) {  // per-wave tile metadata
    int idx = base + lane;
    int cidx = idx < nEr ? idx : nEr - 1;
    int e = bucket[off + cidx];
    int src = (int)eidx[e];
    int dst = (int)eidx[NE + e];
    int c = cnt[dst * RR + r];
    s_src[wave][lane] = src;
    s_dst[wave][lane] = dst;
    s_scl[wave][lane] = (idx < nEr) ? 1.0f / (float)(c > 1 ? c : 1) : 0.0f;
  }
  stage_tile_128x128_f16(Wt + (size_t)r * HH * HH, s_w);  // includes barrier
  if (!active) return;

  const float* row = X + (size_t)s_src[wave][nn] * HH;
  v8f acc[8];
#pragma unroll
  for (int nt = 0; nt < 8; ++nt)
#pragma unroll
    for (int j = 0; j < 8; ++j) acc[nt][j] = 0.f;

#pragma unroll
  for (int ks = 0; ks < 4; ++ks) {
    int k0 = ks * 32;
    v16h a = load_a_frag_f32(row, k0, half);
#pragma unroll
    for (int nt = 0; nt < 8; ++nt) {
      v16h b = load_b_frag(s_w + (size_t)(nt * 16 + nn) * HH + k0 + half * 8);
      acc[nt] = wmma_f16(a, b, acc[nt]);
    }
  }
#pragma unroll
  for (int j = 0; j < 8; ++j) {
    int mi = half * 8 + j;
    float s = s_scl[wave][mi];
    if (s != 0.f) {
      float* yrow = Y + (size_t)s_dst[wave][mi] * HH;
#pragma unroll
      for (int nt = 0; nt < 8; ++nt)
        atomicAdd(&yrow[nt * 16 + nn], acc[nt][j] * s);
    }
  }
}

// 8) elementwise relu
__global__ void k_relu(const float* __restrict__ in, float* __restrict__ out, int n) {
  int i = blockIdx.x * blockDim.x + threadIdx.x;
  if (i < n) out[i] = fmaxf(in[i], 0.f);
}

// ---------------------------------------------------------------------------
// 9) DEDICOM decoder. Per 16-edge tile (one wave):
//    stage u=h[t0], v=h[t1] in LDS; for each r:
//      Z = (v*D_r) @ R^T   (WMMA, Bt[n=i][k=j] = R[i][j] row-major f16)
//      out[e,r] = sigmoid( sum_i u[e,i]*D_r[i]*Z[e,i] )  via LDS ds_add_f32
// ---------------------------------------------------------------------------
__global__ __launch_bounds__(128) void k_dec(
    const float* __restrict__ Hn, const long long* __restrict__ tei,
    const _Float16* __restrict__ Rh, const float* __restrict__ Dm,
    float* __restrict__ out) {
  __shared__ __align__(16) _Float16 s_R[HH * HH];
  __shared__ __align__(16) float s_u[4][16][HH];
  __shared__ __align__(16) float s_v[4][16][HH];
  __shared__ __align__(16) float s_red[4][16];

  int wave = threadIdx.x >> 5, lane = threadIdx.x & 31;
  int half = lane >> 4, nn = lane & 15;
  int base = (blockIdx.x * 4 + wave) * 16;
  bool active = (base < NT);

  if (active) {  // cooperative stage: each lane loads half a row of u and v
    int m = lane >> 1;
    int e = base + m; if (e >= NT) e = NT - 1;
    const float* pu = Hn + (size_t)tei[e] * HH;
    const float* pv = Hn + (size_t)tei[NT + e] * HH;
    int c0 = (lane & 1) * 64;
#pragma unroll
    for (int c = 0; c < 64; c += 4) {
      *(float4*)&s_u[wave][m][c0 + c] = *(const float4*)&pu[c0 + c];
      *(float4*)&s_v[wave][m][c0 + c] = *(const float4*)&pv[c0 + c];
    }
    if (lane < 16) s_red[wave][lane] = 0.f;
  }
#if USE_TDM
  if (threadIdx.x < 32) {
    tdm_issue_128x128_f16(Rh, s_R);
  }
  __syncthreads();
#else
  {
    const float4* src = (const float4*)Rh;
    float4* dst = (float4*)s_R;
    for (int i = threadIdx.x; i < 2048; i += 128) dst[i] = src[i];
    __syncthreads();
  }
#endif
  if (!active) return;

  for (int r = 0; r < RR; ++r) {
    const float* Dr = Dm + r * HH;
    v8f acc[8];
#pragma unroll
    for (int nt = 0; nt < 8; ++nt)
#pragma unroll
      for (int j = 0; j < 8; ++j) acc[nt][j] = 0.f;

#pragma unroll
    for (int ks = 0; ks < 4; ++ks) {
      int k0 = ks * 32;
      v16h a;
#pragma unroll
      for (int part = 0; part < 2; ++part) {
        int kb = k0 + part * 16 + half * 8;
        const float4* pv = (const float4*)&s_v[wave][nn][kb];
        const float4* pd = (const float4*)&Dr[kb];
        float4 v0 = pv[0], v1 = pv[1], d0 = pd[0], d1 = pd[1];
        int t = part * 8;
        a[t+0]=(_Float16)(v0.x*d0.x); a[t+1]=(_Float16)(v0.y*d0.y);
        a[t+2]=(_Float16)(v0.z*d0.z); a[t+3]=(_Float16)(v0.w*d0.w);
        a[t+4]=(_Float16)(v1.x*d1.x); a[t+5]=(_Float16)(v1.y*d1.y);
        a[t+6]=(_Float16)(v1.z*d1.z); a[t+7]=(_Float16)(v1.w*d1.w);
      }
#pragma unroll
      for (int nt = 0; nt < 8; ++nt) {
        v16h b = load_b_frag(s_R + (size_t)(nt * 16 + nn) * HH + k0 + half * 8);
        acc[nt] = wmma_f16(a, b, acc[nt]);
      }
    }
    // epilogue: per-lane partial dot over this lane's 8 columns x 8 rows
    float p[8];
#pragma unroll
    for (int j = 0; j < 8; ++j) p[j] = 0.f;
#pragma unroll
    for (int nt = 0; nt < 8; ++nt) {
      int ng = nt * 16 + nn;
      float dg = Dr[ng];
#pragma unroll
      for (int j = 0; j < 8; ++j)
        p[j] += acc[nt][j] * s_u[wave][half * 8 + j][ng] * dg;
    }
#pragma unroll
    for (int j = 0; j < 8; ++j)
      atomicAdd(&s_red[wave][half * 8 + j], p[j]);   // ds_add_f32
    asm volatile("s_wait_dscnt 0x0" ::: "memory");   // intra-wave LDS ordering
    if (lane < 16) {
      int e = base + lane;
      if (e < NT) {
        float x = s_red[wave][lane];
        out[(size_t)e * RR + r] = 1.0f / (1.0f + __expf(-x));
      }
      s_red[wave][lane] = 0.f;
    }
    asm volatile("s_wait_dscnt 0x0" ::: "memory");
  }
}

// ---------------------------------------------------------------------------
extern "C" void kernel_launch(void* const* d_in, const int* in_sizes, int n_in,
                              void* d_out, int out_size, void* d_ws, size_t ws_size,
                              hipStream_t stream) {
  (void)in_sizes; (void)n_in; (void)out_size; (void)ws_size;
  const float*     x     = (const float*)d_in[0];
  const long long* eidx  = (const long long*)d_in[1];
  const float*     attr  = (const float*)d_in[2];
  const long long* tei   = (const long long*)d_in[3];
  const float*     bas1  = (const float*)d_in[4];
  const float*     cmp1  = (const float*)d_in[5];
  const float*     rt1   = (const float*)d_in[6];
  const float*     bia1  = (const float*)d_in[7];
  const float*     bas2  = (const float*)d_in[8];
  const float*     cmp2  = (const float*)d_in[9];
  const float*     rt2   = (const float*)d_in[10];
  const float*     bia2  = (const float*)d_in[11];
  const float*     Rm    = (const float*)d_in[12];
  const float*     Dm    = (const float*)d_in[13];
  float* out = (float*)d_out;

  char* ws = (char*)d_ws;
  size_t off = 0;
  auto take = [&](size_t b) { size_t r = off; off = (off + b + 255) & ~(size_t)255; return r; };
  int*       etype  = (int*)      (ws + take((size_t)NE * 4));
  int*       cnt    = (int*)      (ws + take((size_t)NN * RR * 4));
  int*       bcnt   = (int*)      (ws + take(RR * 4));
  int*       boff   = (int*)      (ws + take((RR + 1) * 4));
  int*       bfill  = (int*)      (ws + take(RR * 4));
  int*       bucket = (int*)      (ws + take((size_t)NE * 4));
  float*     hpre   = (float*)    (ws + take((size_t)NN * HH * 4));
  float*     h1     = (float*)    (ws + take((size_t)NN * HH * 4));
  float*     h2     = (float*)    (ws + take((size_t)NN * HH * 4));
  _Float16*  W1t    = (_Float16*) (ws + take((size_t)RR * HH * HH * 2));
  _Float16*  W2t    = (_Float16*) (ws + take((size_t)RR * HH * HH * 2));
  _Float16*  r1t    = (_Float16*) (ws + take((size_t)HH * HH * 2));
  _Float16*  r2t    = (_Float16*) (ws + take((size_t)HH * HH * 2));
  _Float16*  Rh     = (_Float16*) (ws + take((size_t)HH * HH * 2));

  (void)hipMemsetAsync(cnt,   0, (size_t)NN * RR * 4, stream);
  (void)hipMemsetAsync(bcnt,  0, RR * 4, stream);
  (void)hipMemsetAsync(bfill, 0, RR * 4, stream);

  // ---- preprocessing: etype/argmax, counts, bucket sort, weight expansion
  k_etype  <<<(NE + 255) / 256, 256, 0, stream>>>(attr, eidx, etype, bcnt, cnt);
  k_scan   <<<1, 32, 0, stream>>>(bcnt, boff);
  k_scatter<<<(NE + 255) / 256, 256, 0, stream>>>(etype, boff, bfill, bucket);
  k_expand_w<<<(RR * HH * HH + 255) / 256, 256, 0, stream>>>(cmp1, bas1, W1t);
  k_expand_w<<<(RR * HH * HH + 255) / 256, 256, 0, stream>>>(cmp2, bas2, W2t);
  k_cvt<<<(HH * HH + 255) / 256, 256, 0, stream>>>(rt1, r1t, 1);
  k_cvt<<<(HH * HH + 255) / 256, 256, 0, stream>>>(rt2, r2t, 1);
  k_cvt<<<(HH * HH + 255) / 256, 256, 0, stream>>>(Rm, Rh, 0);

  const int gemm_blocks = (NN + 127) / 128;                 // 157
  const dim3 rel_grid((((NE + 15) / 16) + 7) / 8, RR);      // (7813, 32)

  // ---- layer 1
  k_dense_gemm<<<gemm_blocks, 256, 0, stream>>>(x, r1t, bia1, hpre, NN);
  k_rel_gemm  <<<rel_grid, 256, 0, stream>>>(x, W1t, eidx, bucket, boff, cnt, hpre);
  k_relu      <<<(NN * HH + 255) / 256, 256, 0, stream>>>(hpre, h1, NN * HH);

  // ---- layer 2
  k_dense_gemm<<<gemm_blocks, 256, 0, stream>>>(h1, r2t, bia2, hpre, NN);
  k_rel_gemm  <<<rel_grid, 256, 0, stream>>>(h1, W2t, eidx, bucket, boff, cnt, hpre);
  k_relu      <<<(NN * HH + 255) / 256, 256, 0, stream>>>(hpre, h2, NN * HH);

  // ---- DEDICOM decoder
  k_dec<<<(((NT + 15) / 16) + 3) / 4, 128, 0, stream>>>(h2, tei, Rh, Dm, out);
}
